// MGDA_15444702396595
// MI455X (gfx1250) — matrix operand order repaired and checked by reading, not compiled, not run on hardware
//
#include <hip/hip_runtime.h>
#include <hip/hip_bf16.h>

typedef _Float16 f16x8  __attribute__((ext_vector_type(8)));
typedef _Float16 f16x16 __attribute__((ext_vector_type(16)));
typedef float    f32x8  __attribute__((ext_vector_type(8)));

#define CAT16(lo,hi) __builtin_shufflevector((lo),(hi),0,1,2,3,4,5,6,7,8,9,10,11,12,13,14,15)
#define WMMA_F16(a,b,c) __builtin_amdgcn_wmma_f32_16x16x32_f16(false,(a),false,(b),(short)0,(c),false,false)

__device__ __forceinline__ float lrelu_f(float x){ return x >= 0.f ? x : 0.01f*x; }

// ---------------- weight packing: OIHW f32 -> [Npad x K] f16 (row n contiguous in k) ------------
__global__ void k_pack_w(const float* __restrict__ w, _Float16* __restrict__ bt,
                         int O, int Opad, int Kw){
  int i = blockIdx.x*blockDim.x + threadIdx.x;
  if (i >= Opad*Kw) return;
  int n = i / Kw, k = i - n*Kw;
  bt[i] = (_Float16)(n < O ? w[(size_t)n*Kw + k] : 0.f);
}

// deconv weight: wt[oc][ic][ky][kx] = w[ic][oc][2-ky][2-kx];  BT[oc][ic*9+ky*3+kx]
__global__ void k_pack_w_deconv(const float* __restrict__ w, _Float16* __restrict__ bt, int C){
  int i = blockIdx.x*blockDim.x + threadIdx.x;
  if (i >= C*C*9) return;
  int oc = i / (C*9); int r = i - oc*C*9; int ic = r/9; int kk = r - ic*9;
  int ky = kk/3, kx = kk - ky*3;
  bt[i] = (_Float16)w[((size_t)ic*C + oc)*9 + (2-ky)*3 + (2-kx)];
}

// ---------------- im2col: NCHW concat(src1,src2) 3x3 conv, k = ic*9 + ky*3+kx ------------------
__global__ void k_im2col_nchw2(const float* __restrict__ s1, const float* __restrict__ s2,
    int C1, int Ct, int H, int W, int OH, int OW, int stride, int pad,
    _Float16* __restrict__ A, int K, int total){
  int i = blockIdx.x*blockDim.x + threadIdx.x;
  if (i >= total) return;
  int m = i / K, k = i - m*K;
  int ic = k / 9, kk = k - ic*9;
  int ohw = OH*OW;
  int b = m / ohw, p = m - b*ohw;
  int oy = p / OW, ox = p - oy*OW;
  int iy = oy*stride + kk/3 - pad;
  int ix = ox*stride + (kk - (kk/3)*3) - pad;
  float v = 0.f;
  if (iy >= 0 && iy < H && ix >= 0 && ix < W) {
    const float* src = (ic < C1) ? s1 + ((size_t)b*C1 + ic)*H*W
                                 : s2 + ((size_t)b*(Ct-C1) + (ic-C1))*H*W;
    v = src[iy*W + ix];
  }
  A[i] = (_Float16)v;
}

// im2col from channel-last activations [B*H*W x C], 1x1 (pure f32->f16 copy, coalesced)
__global__ void k_im2col_cl(const float* __restrict__ src, int C, int H, int W,
    int KH, int pad, _Float16* __restrict__ A, int K, int total){
  int i = blockIdx.x*blockDim.x + threadIdx.x;
  if (i >= total) return;
  int m = i / K, k = i - m*K;
  int kq = KH*KH;
  int ic = k / kq, kk = k - ic*kq;
  int hw = H*W;
  int b = m / hw, p = m - b*hw;
  int oy = p / W, ox = p - oy*W;
  int iy = oy + kk/KH - pad;
  int ix = ox + (kk - (kk/KH)*KH) - pad;
  float v = 0.f;
  if (iy >= 0 && iy < H && ix >= 0 && ix < W)
    v = src[((size_t)b*hw + iy*W + ix)*C + ic];
  A[i] = (_Float16)v;
}

// im2col channel-last 3x3 pad1 stride1: one thread per (pixel, ic), ic fastest ->
// coalesced f32 reads across the wave, 9 contiguous f16 writes per thread.
__global__ void k_im2col_cl3(const float* __restrict__ src, int C, int H, int W,
    _Float16* __restrict__ A, int total /* Mtot*C */){
  int t = blockIdx.x*blockDim.x + threadIdx.x;
  if (t >= total) return;
  int m = t / C, ic = t - m*C;
  int hw = H*W;
  int b = m / hw, p = m - b*hw;
  int oy = p / W, ox = p - oy*W;
  const float* sb = src + (size_t)b*hw*C + ic;
  _Float16* Ao = A + (size_t)m*(C*9) + ic*9;
  #pragma unroll
  for (int ky = 0; ky < 3; ++ky){
    int iy = oy + ky - 1;
    bool vy = (iy >= 0) && (iy < H);
    #pragma unroll
    for (int kx = 0; kx < 3; ++kx){
      int ix = ox + kx - 1;
      float v = 0.f;
      if (vy && ix >= 0 && ix < W)
        v = sb[(size_t)(iy*W + ix)*C];
      Ao[ky*3+kx] = (_Float16)v;
    }
  }
}

// im2col for stride-2 transposed conv (lhs_dilation=2, pad (1,2)), channel-last input.
// one thread per (out pixel, ic), ic fastest (coalesced reads).
__global__ void k_im2col_deconv(const float* __restrict__ src, int C, int IH, int IW,
    _Float16* __restrict__ A, int total /* Mtot*C */){
  int t = blockIdx.x*blockDim.x + threadIdx.x;
  if (t >= total) return;
  int m = t / C, ic = t - m*C;
  int OW = IW*2, ohw = (IH*2)*OW;
  int b = m / ohw, p = m - b*ohw;
  int oy = p / OW, ox = p - oy*OW;
  const float* sb = src + (size_t)b*IH*IW*C + ic;
  _Float16* Ao = A + (size_t)m*(C*9) + ic*9;
  #pragma unroll
  for (int ky = 0; ky < 3; ++ky){
    int dy = oy + ky - 1;
    int iy = dy >> 1;
    bool vy = (dy >= 0) && !(dy & 1) && (iy < IH);
    #pragma unroll
    for (int kx = 0; kx < 3; ++kx){
      int dx = ox + kx - 1;
      int ix = dx >> 1;
      float v = 0.f;
      if (vy && dx >= 0 && !(dx & 1) && ix < IW)
        v = sb[(size_t)(iy*IW + ix)*C];
      Ao[ky*3+kx] = (_Float16)v;
    }
  }
}

// deformable-conv im2col: bilinear gather modulated by sigmoid mask. est layout [Mx x 32].
__global__ void k_im2col_dcn(const float* __restrict__ est, const float* __restrict__ x,
    _Float16* __restrict__ A, int mOff, int CH, int H, int W, int C){
  int t = blockIdx.x*blockDim.x + threadIdx.x;
  if (t >= CH*9) return;
  int mL = t / 9, kk = t - mL*9;
  int mg = mOff + mL;
  int hw = H*W;
  int b = mg / hw, p = mg - b*hw;
  int y = p / W, xq = p - y*W;
  const float* e = est + (size_t)mg*32;
  float mask = 1.f/(1.f + __expf(-e[kk]));
  float py = (float)(y + kk/3 - 1) + e[9 + 2*kk];
  float px = (float)(xq + (kk - (kk/3)*3) - 1) + e[10 + 2*kk];
  float y0f = floorf(py), x0f = floorf(px);
  int y0 = (int)y0f, x0 = (int)x0f;
  float wy1 = py - y0f, wx1 = px - x0f;
  bool vy0 = (y0 >= 0) & (y0 < H),   vy1 = (y0+1 >= 0) & (y0+1 < H);
  bool vx0 = (x0 >= 0) & (x0 < W),   vx1 = (x0+1 >= 0) & (x0+1 < W);
  float w00 = (vy0&&vx0) ? (1.f-wy1)*(1.f-wx1) : 0.f;
  float w01 = (vy0&&vx1) ? (1.f-wy1)*wx1       : 0.f;
  float w10 = (vy1&&vx0) ? wy1*(1.f-wx1)       : 0.f;
  float w11 = (vy1&&vx1) ? wy1*wx1             : 0.f;
  int yc0 = min(max(y0,0),H-1),   yc1 = min(max(y0+1,0),H-1);
  int xc0 = min(max(x0,0),W-1),   xc1 = min(max(x0+1,0),W-1);
  int i00 = yc0*W+xc0, i01 = yc0*W+xc1, i10 = yc1*W+xc0, i11 = yc1*W+xc1;
  const float* xb = x + (size_t)b*C*hw;
  _Float16* Ao = A + (size_t)mL*(C*9) + kk;
  for (int ic = 0; ic < C; ++ic) {
    const float* xc = xb + (size_t)ic*hw;
    float v = w00*xc[i00] + w01*xc[i01] + w10*xc[i10] + w11*xc[i11];
    Ao[ic*9] = (_Float16)(mask * v);
  }
}

// ---------------- generic f16 WMMA GEMM: C[MxN] = A[MxK] x BT[NxK]^T, fused epilogue ----------
__global__ __launch_bounds__(256) void k_gemm(
    const _Float16* __restrict__ A, const _Float16* __restrict__ BT,
    const float* __restrict__ bias, int Nb,
    const float* __restrict__ res1, float s1,
    const float* __restrict__ res2, float s2,
    int act,
    float* __restrict__ outF32, _Float16* __restrict__ outF16,
    _Float16* __restrict__ outF16T, int ldT,
    float* __restrict__ outNCHW, int HW,
    int M, int N, int K, int mOff)
{
  int wid  = (blockIdx.x * blockDim.x + threadIdx.x) >> 5;
  int lane = threadIdx.x & 31;
  int tilesN = N >> 4;
  int tiles  = (M >> 4) * tilesN;
  if (wid >= tiles) return;
  int tm = (wid / tilesN) << 4;
  int tn = (wid - (wid/tilesN)*tilesN) << 4;
  int hh = lane >> 4, li = lane & 15;
  const _Float16* Ap = A  + (size_t)(tm + li) * K;
  const _Float16* Bp = BT + (size_t)(tn + li) * K;
  f32x8 acc = {0.f,0.f,0.f,0.f,0.f,0.f,0.f,0.f};
  for (int k = 0; k < K; k += 32) {
    // stream-ahead prefetch (gfx1250 global_prefetch_b8); OOB speculative prefetch is dropped
    __builtin_prefetch(Ap + k + 512, 0, 1);
    __builtin_prefetch(Bp + k + 512, 0, 1);
    f16x8 a0 = *(const f16x8*)(Ap + k + hh*8);
    f16x8 a1 = *(const f16x8*)(Ap + k + 16 + hh*8);
    f16x8 b0 = *(const f16x8*)(Bp + k + hh*8);
    f16x8 b1 = *(const f16x8*)(Bp + k + 16 + hh*8);
    acc = WMMA_F16(CAT16(a0,a1), CAT16(b0,b1), acc);
  }
  int n = tn + li;
  float bn = (bias && n < Nb) ? bias[n] : 0.f;
  #pragma unroll
  for (int r = 0; r < 8; ++r) {
    int mg = tm + r + 8*hh + mOff;
    float v = acc[r] + bn;
    if (res1) v += s1 * res1[(size_t)mg*N + n];
    if (res2) v += s2 * res2[(size_t)mg*N + n];
    if (act)  v = lrelu_f(v);
    if (outF32)  outF32[(size_t)mg*N + n] = v;
    if (outF16)  outF16[(size_t)mg*N + n] = (_Float16)v;
    if (outF16T) outF16T[(size_t)n*ldT + mg] = (_Float16)v;
    if (outNCHW) {
      int b = mg / HW, pix = mg - b*HW;
      outNCHW[((size_t)b*N + n)*HW + pix] = v;
    }
  }
}

// ---------------- fused non-local attention (flash style), d=32, one wave per 16 rows ---------
__global__ __launch_bounds__(32) void k_attn(
    const _Float16* __restrict__ th, const _Float16* __restrict__ ph,
    const _Float16* __restrict__ gT, _Float16* __restrict__ yOut,
    int Npix, int Mtot)
{
  __shared__ _Float16 lds[16*32];
  int wid = blockIdx.x;
  int lane = threadIdx.x;
  int hh = lane >> 4, li = lane & 15;
  int tilesPerB = Npix >> 4;
  int b = wid / tilesPerB;
  int nBase = (wid - b*tilesPerB) << 4;
  size_t base = (size_t)b * Npix;

  const _Float16* tp = th + (base + nBase + li) * 32;
  f16x16 aT = CAT16(*(const f16x8*)(tp + hh*8), *(const f16x8*)(tp + 16 + hh*8));

  float rowMax[8], rowSum[8];
  f32x8 y0 = {0.f,0.f,0.f,0.f,0.f,0.f,0.f,0.f};
  f32x8 y1 = {0.f,0.f,0.f,0.f,0.f,0.f,0.f,0.f};
  #pragma unroll
  for (int r = 0; r < 8; ++r){ rowMax[r] = -3.0e38f; rowSum[r] = 0.f; }

  for (int mB = 0; mB < Npix; mB += 32) {
    const _Float16* pp0 = ph + (base + mB + li) * 32;
    const _Float16* pp1 = ph + (base + mB + 16 + li) * 32;
    f16x16 bP0 = CAT16(*(const f16x8*)(pp0 + hh*8), *(const f16x8*)(pp0 + 16 + hh*8));
    f16x16 bP1 = CAT16(*(const f16x8*)(pp1 + hh*8), *(const f16x8*)(pp1 + 16 + hh*8));
    f32x8 z = {0.f,0.f,0.f,0.f,0.f,0.f,0.f,0.f};
    f32x8 s0 = WMMA_F16(aT, bP0, z);
    f32x8 s1 = WMMA_F16(aT, bP1, z);
    float p0[8], p1[8];
    #pragma unroll
    for (int r = 0; r < 8; ++r) {
      float v = fmaxf(s0[r], s1[r]);
      v = fmaxf(v, __shfl_xor(v, 1, 32));
      v = fmaxf(v, __shfl_xor(v, 2, 32));
      v = fmaxf(v, __shfl_xor(v, 4, 32));
      v = fmaxf(v, __shfl_xor(v, 8, 32));
      float nm = fmaxf(rowMax[r], v);
      float sc = __expf(rowMax[r] - nm);
      float e0 = __expf(s0[r] - nm);
      float e1 = __expf(s1[r] - nm);
      float ls = e0 + e1;
      ls += __shfl_xor(ls, 1, 32);
      ls += __shfl_xor(ls, 2, 32);
      ls += __shfl_xor(ls, 4, 32);
      ls += __shfl_xor(ls, 8, 32);
      rowSum[r] = rowSum[r]*sc + ls;
      rowMax[r] = nm;
      y0[r] *= sc; y1[r] *= sc;
      p0[r] = e0; p1[r] = e1;
    }
    #pragma unroll
    for (int r = 0; r < 8; ++r) {
      int rr = r + 8*hh;
      lds[rr*32 + li]      = (_Float16)p0[r];
      lds[rr*32 + 16 + li] = (_Float16)p1[r];
    }
    __syncthreads();
    const _Float16* lp = &lds[0] + li*32;
    f16x16 aP = CAT16(*(const f16x8*)(lp + hh*8), *(const f16x8*)(lp + 16 + hh*8));
    const _Float16* g0p = gT + (size_t)li*Mtot + base + mB + hh*8;
    const _Float16* g1p = gT + (size_t)(li+16)*Mtot + base + mB + hh*8;
    f16x16 bG0 = CAT16(*(const f16x8*)g0p, *(const f16x8*)(g0p+16));
    f16x16 bG1 = CAT16(*(const f16x8*)g1p, *(const f16x8*)(g1p+16));
    y0 = WMMA_F16(aP, bG0, y0);
    y1 = WMMA_F16(aP, bG1, y1);
    __syncthreads();
  }
  #pragma unroll
  for (int r = 0; r < 8; ++r) {
    int n = nBase + r + 8*hh;
    float inv = 1.f / rowSum[r];
    _Float16* o = yOut + (base + n)*32;
    o[li]      = (_Float16)(y0[r]*inv);
    o[16 + li] = (_Float16)(y1[r]*inv);
  }
}

// ====================================================================================
extern "C" void kernel_launch(void* const* d_in, const int* in_sizes, int n_in,
                              void* d_out, int out_size, void* d_ws, size_t ws_size,
                              hipStream_t stream) {
  (void)in_sizes; (void)n_in; (void)out_size; (void)ws_size;
  const float* x_all   = (const float*)d_in[0];
  const float* ms_fine = (const float*)d_in[1];
  const float* ms_co   = (const float*)d_in[2];
  const float* enc_w0  = (const float*)d_in[3];
  const float* enc_b0  = (const float*)d_in[4];
  const float* enc_w1  = (const float*)d_in[5];
  const float* enc_b1  = (const float*)d_in[6];
  const float* nl_tw[2] = {(const float*)d_in[7],  (const float*)d_in[15]};
  const float* nl_tb[2] = {(const float*)d_in[8],  (const float*)d_in[16]};
  const float* nl_pw[2] = {(const float*)d_in[9],  (const float*)d_in[17]};
  const float* nl_pb[2] = {(const float*)d_in[10], (const float*)d_in[18]};
  const float* nl_gw[2] = {(const float*)d_in[11], (const float*)d_in[19]};
  const float* nl_gb[2] = {(const float*)d_in[12], (const float*)d_in[20]};
  const float* nl_ww[2] = {(const float*)d_in[13], (const float*)d_in[21]};
  const float* nl_wb[2] = {(const float*)d_in[14], (const float*)d_in[22]};
  const float* dec_w0  = (const float*)d_in[23];
  const float* dec_b0  = (const float*)d_in[24];
  const float* dec_w1  = (const float*)d_in[25];
  const float* dec_b1  = (const float*)d_in[26];
  const float* off_w   = (const float*)d_in[27];
  const float* off_b   = (const float*)d_in[28];
  const float* dcn_w   = (const float*)d_in[29];
  const float* dcn_b   = (const float*)d_in[30];
  float* out = (float*)d_out;

  // ---- workspace carve (all 256B aligned) ----
  char* ws = (char*)d_ws;
  size_t off = 0;
  auto carve = [&](size_t bytes)->char*{
    char* p = ws + off;
    off = (off + bytes + 255) & ~(size_t)255;
    return p;
  };
  _Float16* bt_enc0 = (_Float16*)carve((size_t)64*4608*2);
  _Float16* bt_enc1 = (_Float16*)carve((size_t)64*2304*2);
  _Float16 *bt_t[2], *bt_p[2], *bt_g[2], *bt_w[2];
  for (int i = 0; i < 2; ++i) {
    bt_t[i] = (_Float16*)carve(32*64*2);
    bt_p[i] = (_Float16*)carve(32*64*2);
    bt_g[i] = (_Float16*)carve(32*64*2);
    bt_w[i] = (_Float16*)carve(64*32*2);
  }
  _Float16* bt_dec0 = (_Float16*)carve((size_t)64*576*2);
  _Float16* bt_dec1 = (_Float16*)carve((size_t)64*576*2);
  _Float16* bt_off  = (_Float16*)carve((size_t)32*576*2);
  _Float16* bt_dcn  = (_Float16*)carve((size_t)128*1152*2);
  _Float16* Abuf    = (_Float16*)carve((size_t)32768*576*2);    // 37.75MB shared im2col buffer
  float*    e0  = (float*)carve((size_t)2048*64*4);
  _Float16* th0 = (_Float16*)carve((size_t)2048*32*2);
  _Float16* ph0 = (_Float16*)carve((size_t)2048*32*2);
  _Float16* gT0 = (_Float16*)carve((size_t)2048*32*2);
  _Float16* yv0 = (_Float16*)carve((size_t)2048*32*2);
  float*    m0  = (float*)carve((size_t)2048*64*4);
  float*    u0  = (float*)carve((size_t)8192*64*4);
  float*    e1  = (float*)carve((size_t)8192*64*4);
  _Float16* th1 = (_Float16*)carve((size_t)8192*32*2);
  _Float16* ph1 = (_Float16*)carve((size_t)8192*32*2);
  _Float16* gT1 = (_Float16*)carve((size_t)8192*32*2);
  _Float16* yv1 = (_Float16*)carve((size_t)8192*32*2);
  float*    mm1 = (float*)carve((size_t)8192*64*4);
  float*    mot = (float*)carve((size_t)32768*64*4);
  float*    est = (float*)carve((size_t)32768*32*4);

  auto nb = [](long long t){ return (unsigned)((t + 255)/256); };

  // ---- pack all weights to f16 [Npad x K] ----
  k_pack_w<<<nb(64*4608),256,0,stream>>>(enc_w0, bt_enc0, 64,64,4608);
  k_pack_w<<<nb(64*2304),256,0,stream>>>(enc_w1, bt_enc1, 64,64,2304);
  for (int i = 0; i < 2; ++i) {
    k_pack_w<<<nb(32*64),256,0,stream>>>(nl_tw[i], bt_t[i], 32,32,64);
    k_pack_w<<<nb(32*64),256,0,stream>>>(nl_pw[i], bt_p[i], 32,32,64);
    k_pack_w<<<nb(32*64),256,0,stream>>>(nl_gw[i], bt_g[i], 32,32,64);
    k_pack_w<<<nb(64*32),256,0,stream>>>(nl_ww[i], bt_w[i], 64,64,32);
  }
  k_pack_w_deconv<<<nb(64*64*9),256,0,stream>>>(dec_w0, bt_dec0, 64);
  k_pack_w_deconv<<<nb(64*64*9),256,0,stream>>>(dec_w1, bt_dec1, 64);
  k_pack_w<<<nb(32*576),256,0,stream>>>(off_w, bt_off, 27,32,576);
  k_pack_w<<<nb(128*1152),256,0,stream>>>(dcn_w, bt_dcn, 128,128,1152);

  // aligned[0] = x_all[0]
  hipMemcpyAsync(out, x_all, (size_t)2*128*16384*4, hipMemcpyDeviceToDevice, stream);

  auto gemm = [&](const _Float16* A, const _Float16* BT, const float* bias, int Nb,
                  const float* r1, float s1, const float* r2, float s2, int act,
                  float* oF, _Float16* oH, _Float16* oHT, int ldT,
                  float* oN, int HW, int M, int N, int K, int mOff){
    int tiles = (M/16)*(N/16);
    k_gemm<<<(tiles+7)/8, 256, 0, stream>>>(A,BT,bias,Nb,r1,s1,r2,s2,act,
                                            oF,oH,oHT,ldT,oN,HW,M,N,K,mOff);
  };

  for (int s = 1; s <= 2; ++s) {
    const float* co_s = ms_co   + (size_t)s*2*256*64*64;
    const float* fi_s = ms_fine + (size_t)s*2*128*16384;
    const float* xs   = x_all   + (size_t)s*2*128*16384;
    float* outS = out + (size_t)s*2*128*16384;

    // ---- coarse branch: enc0 -> nonlocal0 -> deconv0 ----
    k_im2col_nchw2<<<nb((long long)2048*4608),256,0,stream>>>(co_s, ms_co, 256,512, 64,64,32,32, 2,1, Abuf, 4608, 2048*4608);
    gemm(Abuf, bt_enc0, enc_b0,64, nullptr,0,nullptr,0, 1, e0,nullptr,nullptr,0, nullptr,0, 2048,64,4608,0);
    k_im2col_cl<<<nb(2048*64),256,0,stream>>>(e0, 64, 32,32, 1,0, Abuf, 64, 2048*64);
    gemm(Abuf, bt_t[0], nl_tb[0],32, nullptr,0,nullptr,0, 0, nullptr,th0,nullptr,0, nullptr,0, 2048,32,64,0);
    gemm(Abuf, bt_p[0], nl_pb[0],32, nullptr,0,nullptr,0, 0, nullptr,ph0,nullptr,0, nullptr,0, 2048,32,64,0);
    gemm(Abuf, bt_g[0], nl_gb[0],32, nullptr,0,nullptr,0, 0, nullptr,nullptr,gT0,2048, nullptr,0, 2048,32,64,0);
    k_attn<<<2*1024/16, 32, 0, stream>>>(th0, ph0, gT0, yv0, 1024, 2048);
    gemm(yv0, bt_w[0], nl_wb[0],64, e0,2.0f,nullptr,0, 0, m0,nullptr,nullptr,0, nullptr,0, 2048,64,32,0);   // m0 = 2*e0 + W(y)+b
    k_im2col_deconv<<<nb((long long)8192*64),256,0,stream>>>(m0, 64, 32,32, Abuf, 8192*64);
    gemm(Abuf, bt_dec0, dec_b0,64, nullptr,0,nullptr,0, 1, u0,nullptr,nullptr,0, nullptr,0, 8192,64,576,0);

    // ---- fine branch: enc1 -> nonlocal1 (+u0) -> deconv1 ----
    k_im2col_nchw2<<<nb((long long)8192*2304),256,0,stream>>>(fi_s, ms_fine, 128,256, 128,128,64,64, 2,1, Abuf, 2304, 8192*2304);
    gemm(Abuf, bt_enc1, enc_b1,64, nullptr,0,nullptr,0, 1, e1,nullptr,nullptr,0, nullptr,0, 8192,64,2304,0);
    k_im2col_cl<<<nb(8192*64),256,0,stream>>>(e1, 64, 64,64, 1,0, Abuf, 64, 8192*64);
    gemm(Abuf, bt_t[1], nl_tb[1],32, nullptr,0,nullptr,0, 0, nullptr,th1,nullptr,0, nullptr,0, 8192,32,64,0);
    gemm(Abuf, bt_p[1], nl_pb[1],32, nullptr,0,nullptr,0, 0, nullptr,ph1,nullptr,0, nullptr,0, 8192,32,64,0);
    gemm(Abuf, bt_g[1], nl_gb[1],32, nullptr,0,nullptr,0, 0, nullptr,nullptr,gT1,8192, nullptr,0, 8192,32,64,0);
    k_attn<<<2*4096/16, 32, 0, stream>>>(th1, ph1, gT1, yv1, 4096, 8192);
    gemm(yv1, bt_w[1], nl_wb[1],64, e1,2.0f,u0,1.0f, 0, mm1,nullptr,nullptr,0, nullptr,0, 8192,64,32,0);    // mm1 = m1 + u0
    k_im2col_deconv<<<nb((long long)32768*64),256,0,stream>>>(mm1, 64, 64,64, Abuf, 32768*64);
    gemm(Abuf, bt_dec1, dec_b1,64, nullptr,0,nullptr,0, 1, mot,nullptr,nullptr,0, nullptr,0, 32768,64,576,0);

    // ---- offset/mask conv -> deformable conv (chunked), scatter to NCHW output ----
    k_im2col_cl3<<<nb((long long)32768*64),256,0,stream>>>(mot, 64, 128,128, Abuf, 32768*64);
    gemm(Abuf, bt_off, off_b,27, nullptr,0,nullptr,0, 0, est,nullptr,nullptr,0, nullptr,0, 32768,32,576,0);
    for (int c0 = 0; c0 < 32768; c0 += 8192) {
      k_im2col_dcn<<<nb(8192*9),256,0,stream>>>(est, xs, Abuf, c0, 8192, 128,128, 128);
      gemm(Abuf, bt_dcn, dcn_b,128, nullptr,0,nullptr,0, 0, nullptr,nullptr,nullptr,0, outS,16384, 8192,128,1152, c0);
    }
  }
}
